// VFPG_ours_87935160419018
// MI455X (gfx1250) — matrix operand order repaired and checked by reading, not compiled, not run on hardware
//
#include <hip/hip_runtime.h>
#include <hip/hip_bf16.h>

typedef __attribute__((ext_vector_type(16))) _Float16 v16h;
typedef __attribute__((ext_vector_type(8)))  _Float16 v8h;
typedef __attribute__((ext_vector_type(8)))  float    v8f;

#define M_TOT   4096
#define HID     512
#define GH      2048      // 4*HID
#define NCOMP   32
#define OUTC    96
#define NSTEPS  62
#define HPAD    520       // 16B-aligned padded row (520*2 = 1040 = 65*16)
#define KT      16        // HID / 32

#if __has_builtin(__builtin_amdgcn_sched_barrier)
#define SCHED_FENCE() __builtin_amdgcn_sched_barrier(0)
#else
#define SCHED_FENCE() asm volatile("" ::: "memory")
#endif

// ---------------------------------------------------------------------------
// Prep kernels
// ---------------------------------------------------------------------------

// Repack a (nrows x 512) fp32 weight into fragment-major f16 matching the
// CDNA5 WMMA B-matrix (32x16 f16) per-lane layout:
//   frag(nt,kt): lane L holds column n = nt*16 + (L%16),
//                K values kt*32 + (L/16)*8 .. +7, then kt*32 + 16 + (L/16)*8 .. +7.
__global__ void pack_w16(const float* __restrict__ src, _Float16* __restrict__ dst,
                         int nrows) {
    int tid   = blockIdx.x * blockDim.x + threadIdx.x;
    int total = (nrows / 16) * KT * 32;
    if (tid >= total) return;
    int lane = tid & 31;
    int kt   = (tid >> 5) & (KT - 1);
    int nt   = tid >> 9;
    int row  = nt * 16 + (lane & 15);
    int kb   = kt * 32 + (lane >> 4) * 8;
    const float* s = src + (size_t)row * HID + kb;
    _Float16*    d = dst + ((size_t)tid << 4);
#pragma unroll
    for (int e = 0; e < 16; ++e) d[e] = (_Float16)s[e + (e & 8)];
}

__global__ void bias_comb(const float* __restrict__ a, const float* __restrict__ b,
                          float* __restrict__ dst, int n) {
    int i = blockIdx.x * blockDim.x + threadIdx.x;
    if (i < n) dst[i] = a[i] + b[i];
}

// ---------------------------------------------------------------------------
// Device helpers
// ---------------------------------------------------------------------------

// Branchless fast activations: v_exp_f32 + v_rcp_f32, no divergence.
__device__ __forceinline__ float sigf(float x) {
    return __builtin_amdgcn_rcpf(1.0f + __expf(-x));
}
__device__ __forceinline__ float tanh_fast(float x) {
    // tanh(x) = 1 - 2/(exp(2x)+1)
    return 1.0f - 2.0f * __builtin_amdgcn_rcpf(__expf(2.0f * x) + 1.0f);
}

// A-fragment (16x32 f16) from LDS h-tile, per ISA layout:
// lane L: row r=L%16; halves [hi*8 .. hi*8+7] -> elems 0..7,
//                     halves [16+hi*8 .. +7]  -> elems 8..15  (hi = L/16)
__device__ __forceinline__ v16h load_a_frag(const _Float16* hbuf, int r, int hi, int kt) {
    const _Float16* ap = hbuf + r * HPAD + kt * 32 + hi * 8;
    v8h lo = *(const v8h*)ap;
    v8h hh = *(const v8h*)(ap + 16);
    return __builtin_shufflevector(lo, hh, 0,1,2,3,4,5,6,7,8,9,10,11,12,13,14,15);
}

// GEMM accumulate for ONE hidden n-tile across all four gate blocks:
//   acc[g] += A(h, 16x512) * B(W, n-tile g*32+ntj), g = i,f,g,o
// Only 4 accumulator tiles (32 VGPRs) live -> no spills.
__device__ __forceinline__ void gemm_gates(v8f (&acc)[4], const _Float16* hbuf,
                                           const v16h* __restrict__ wfrag,
                                           int ntj, int lane) {
    int r = lane & 15, hi = lane >> 4;
#pragma unroll 2
    for (int kt = 0; kt < KT; ++kt) {
        v16h a = load_a_frag(hbuf, r, hi, kt);
#pragma unroll
        for (int g = 0; g < 4; ++g) {
            v16h b = wfrag[((size_t)(g * 32 + ntj) * KT + kt) * 32 + lane];
            acc[g] = __builtin_amdgcn_wmma_f32_16x16x32_f16(
                false, a, false, b, (short)0, acc[g], false, false);
        }
    }
}

// ---------------------------------------------------------------------------
// Main persistent kernel: 1 block = 16 batch rows over all 62 steps.
// ---------------------------------------------------------------------------

__launch_bounds__(256)
__global__ void vfpg_main(const float* __restrict__ x_prev,
                          const float* __restrict__ W_ih0,   // (2048,1)
                          const float* __restrict__ b_fc,
                          const float* __restrict__ noise_x, // (61,4096,1)
                          const float* __restrict__ gumbel,  // (62,4096,32)
                          const float* __restrict__ eps,     // (62,4096,1)
                          const _Float16* __restrict__ Whh0p,
                          const _Float16* __restrict__ Wih1p,
                          const _Float16* __restrict__ Whh1p,
                          const _Float16* __restrict__ Wfcp,
                          const float* __restrict__ bc0,     // b_ih0+b_hh0
                          const float* __restrict__ bc1,     // b_ih1+b_hh1
                          float* __restrict__ out) {
    __shared__ _Float16 h0buf[16 * HPAD];
    __shared__ _Float16 h1buf[16 * HPAD];
    __shared__ float    xt[16];
    __shared__ float    ybuf[16][104];

    const int tid    = threadIdx.x;
    const int lane   = tid & 31;
    const int wv     = tid >> 5;          // wave id 0..7
    const int r      = lane & 15;
    const int hi     = lane >> 4;
    const int m_base = blockIdx.x * 16;

    const v16h* Whh0v = (const v16h*)Whh0p;
    const v16h* Wih1v = (const v16h*)Wih1p;
    const v16h* Whh1v = (const v16h*)Whh1p;

    // h0 = h1 = 0
    for (int i = tid; i < 16 * HPAD; i += 256) {
        h0buf[i] = (_Float16)0.f;
        h1buf[i] = (_Float16)0.f;
    }

    v8f c0[4], c1[4];
#pragma unroll
    for (int jt = 0; jt < 4; ++jt)
#pragma unroll
        for (int v = 0; v < 8; ++v) { c0[jt][v] = 0.f; c1[jt][v] = 0.f; }

    __syncthreads();

    for (int step = 0; step < NSTEPS; ++step) {
        if (tid < 16)
            xt[tid] = (step == 0) ? x_prev[m_base + tid]
                                  : noise_x[(size_t)(step - 1) * M_TOT + m_base + tid];
        __syncthreads();

        float xv[8];
#pragma unroll
        for (int v = 0; v < 8; ++v) xv[v] = xt[v + 8 * hi];  // C-layout row m = v+8*hi

        // ------------------ layer 0: one j-tile at a time ------------------
        v8f h0new[4];
#pragma unroll
        for (int jt = 0; jt < 4; ++jt) {
            int ntj = wv * 4 + jt;
            v8f acc[4];
#pragma unroll
            for (int g = 0; g < 4; ++g) {
                int col   = g * HID + ntj * 16 + r;
                float wih = W_ih0[col];              // IN==1: x@Wih.T = outer product
                float bc  = bc0[col];
#pragma unroll
                for (int v = 0; v < 8; ++v) acc[g][v] = xv[v] * wih + bc;
            }
            gemm_gates(acc, h0buf, Whh0v, ntj, lane);
#pragma unroll
            for (int v = 0; v < 8; ++v) {
                float i_ = sigf(acc[0][v]);
                float f_ = sigf(acc[1][v]);
                float g_ = tanh_fast(acc[2][v]);
                float o_ = sigf(acc[3][v]);
                float c2 = f_ * c0[jt][v] + i_ * g_;
                c0[jt][v]   = c2;
                h0new[jt][v] = o_ * tanh_fast(c2);   // defer LDS write past barrier
            }
            SCHED_FENCE();   // keep j-tile blocks from interleaving (reg pressure)
        }
        __syncthreads();  // everyone finished reading old h0
#pragma unroll
        for (int jt = 0; jt < 4; ++jt) {
            int col = (wv * 4 + jt) * 16 + r;
#pragma unroll
            for (int v = 0; v < 8; ++v)
                h0buf[(v + 8 * hi) * HPAD + col] = (_Float16)h0new[jt][v];
        }
        __syncthreads();  // new h0 visible

        // ------------------ layer 1: one j-tile at a time ------------------
        v8f h1new[4];
#pragma unroll
        for (int jt = 0; jt < 4; ++jt) {
            int ntj = wv * 4 + jt;
            v8f acc[4];
#pragma unroll
            for (int g = 0; g < 4; ++g) {
                float bc = bc1[g * HID + ntj * 16 + r];
#pragma unroll
                for (int v = 0; v < 8; ++v) acc[g][v] = bc;
            }
            gemm_gates(acc, h0buf, Wih1v, ntj, lane);  // input = new h0
            gemm_gates(acc, h1buf, Whh1v, ntj, lane);  // recurrent = old h1
#pragma unroll
            for (int v = 0; v < 8; ++v) {
                float i_ = sigf(acc[0][v]);
                float f_ = sigf(acc[1][v]);
                float g_ = tanh_fast(acc[2][v]);
                float o_ = sigf(acc[3][v]);
                float c2 = f_ * c1[jt][v] + i_ * g_;
                c1[jt][v]   = c2;
                h1new[jt][v] = o_ * tanh_fast(c2);
            }
            SCHED_FENCE();
        }
        __syncthreads();  // everyone finished reading old h1
#pragma unroll
        for (int jt = 0; jt < 4; ++jt) {
            int col = (wv * 4 + jt) * 16 + r;
#pragma unroll
            for (int v = 0; v < 8; ++v)
                h1buf[(v + 8 * hi) * HPAD + col] = (_Float16)h1new[jt][v];
        }
        __syncthreads();  // new h1 visible

        // ------------------ FC head (96 cols = 6 n-tiles, waves 0..5) -------
        if (wv < 6) {
            int col = wv * 16 + r;
            v8f y;
            float b = b_fc[col];
#pragma unroll
            for (int v = 0; v < 8; ++v) y[v] = b;
#pragma unroll 2
            for (int kt = 0; kt < KT; ++kt) {
                v16h a  = load_a_frag(h1buf, r, hi, kt);
                v16h bf = ((const v16h*)Wfcp)[((size_t)wv * KT + kt) * 32 + lane];
                y = __builtin_amdgcn_wmma_f32_16x16x32_f16(false, a, false, bf,
                                                           (short)0, y, false, false);
            }
#pragma unroll
            for (int v = 0; v < 8; ++v) ybuf[v + 8 * hi][col] = y[v];
        }
        __syncthreads();

        // ------------------ MDN tail: one thread per batch row --------------
        if (tid < 16) {
            int m = m_base + tid;
            const float* y = ybuf[tid];
            // stable softmax over logits y[0..31]
            float mx = y[0];
            for (int j = 1; j < NCOMP; ++j) mx = fmaxf(mx, y[j]);
            float e[NCOMP], s = 0.f;
            for (int j = 0; j < NCOMP; ++j) { e[j] = __expf(y[j] - mx); s += e[j]; }
            float inv = __builtin_amdgcn_rcpf(s);
            // argmax(log softmax + gumbel) == argmax(y + gumbel)
            const float* gm = gumbel + ((size_t)step * M_TOT + m) * NCOMP;
            int   best = 0;
            float bv   = y[0] + gm[0];
            for (int j = 1; j < NCOMP; ++j) {
                float t = y[j] + gm[j];
                if (t > bv) { bv = t; best = j; }
            }
            float ep = eps[(size_t)step * M_TOT + m];
            float xp = y[2 * NCOMP + best] + __expf(y[NCOMP + best]) * ep;
            // prob = sum_j gamma_j * exp(-0.5 d^2 - 66*log_sig_j - 32*ln(2pi))
            float prob = 0.f;
            for (int j = 0; j < NCOMP; ++j) {
                float d    = xp - y[2 * NCOMP + j];
                float lsig = y[NCOMP + j];
                prob += (e[j] * inv) * __expf(-0.5f * d * d - 66.0f * lsig - 58.8120661f);
            }
            out[(size_t)m * 64 + 1 + step]                      = xp;
            out[(size_t)M_TOT * 64 + (size_t)m * 64 + 1 + step] = prob;
            if (step == 0) {
                float x0 = x_prev[m];
                out[(size_t)m * 64 + 0]                       = x0;
                out[(size_t)m * 64 + 63]                      = x0;
                out[(size_t)M_TOT * 64 + (size_t)m * 64]      = 1.0f;
                out[(size_t)M_TOT * 64 + (size_t)m * 64 + 63] = 1.0f;
            }
        }
        __syncthreads();  // protect ybuf / xt reuse next step
    }
}

// ---------------------------------------------------------------------------
// Launch
// ---------------------------------------------------------------------------

extern "C" void kernel_launch(void* const* d_in, const int* in_sizes, int n_in,
                              void* d_out, int out_size, void* d_ws, size_t ws_size,
                              hipStream_t stream) {
    const float* x_prev  = (const float*)d_in[0];
    const float* W_ih0   = (const float*)d_in[1];
    const float* W_hh0   = (const float*)d_in[2];
    const float* b_ih0   = (const float*)d_in[3];
    const float* b_hh0   = (const float*)d_in[4];
    const float* W_ih1   = (const float*)d_in[5];
    const float* W_hh1   = (const float*)d_in[6];
    const float* b_ih1   = (const float*)d_in[7];
    const float* b_hh1   = (const float*)d_in[8];
    const float* W_fc    = (const float*)d_in[9];
    const float* b_fc    = (const float*)d_in[10];
    const float* noise_x = (const float*)d_in[11];
    const float* gumbel  = (const float*)d_in[12];
    const float* eps     = (const float*)d_in[13];
    float* out           = (float*)d_out;

    // workspace layout
    char* ws = (char*)d_ws;
    _Float16* Whh0p = (_Float16*)(ws + 0);                       // 2 MiB
    _Float16* Wih1p = (_Float16*)(ws + 2u * 1024 * 1024);        // 2 MiB
    _Float16* Whh1p = (_Float16*)(ws + 4u * 1024 * 1024);        // 2 MiB
    _Float16* Wfcp  = (_Float16*)(ws + 6u * 1024 * 1024);        // 96 KiB
    float*    bc0   = (float*)(ws + 6u * 1024 * 1024 + 128u * 1024);
    float*    bc1   = bc0 + GH;

    bias_comb<<<(GH + 255) / 256, 256, 0, stream>>>(b_ih0, b_hh0, bc0, GH);
    bias_comb<<<(GH + 255) / 256, 256, 0, stream>>>(b_ih1, b_hh1, bc1, GH);

    const int big = (GH / 16) * KT * 32;   // 65536 threads
    pack_w16<<<(big + 255) / 256, 256, 0, stream>>>(W_hh0, Whh0p, GH);
    pack_w16<<<(big + 255) / 256, 256, 0, stream>>>(W_ih1, Wih1p, GH);
    pack_w16<<<(big + 255) / 256, 256, 0, stream>>>(W_hh1, Whh1p, GH);
    const int fcn = (OUTC / 16) * KT * 32; // 3072 threads
    pack_w16<<<(fcn + 255) / 256, 256, 0, stream>>>(W_fc, Wfcp, OUTC);

    vfpg_main<<<M_TOT / 16, 256, 0, stream>>>(x_prev, W_ih0, b_fc, noise_x, gumbel,
                                              eps, Whh0p, Wih1p, Whh1p, Wfcp,
                                              bc0, bc1, out);
}
